// TransformerDecoder_35897336660588
// MI455X (gfx1250) — matrix operand order repaired and checked by reading, not compiled, not run on hardware
//
#include <hip/hip_runtime.h>
#include <hip/hip_bf16.h>

// ---------------------------------------------------------------------------
// Types for CDNA5 WMMA (wave32): v16bf A/B fragments, v8f accumulator.
// ---------------------------------------------------------------------------
typedef __attribute__((ext_vector_type(16))) __bf16 v16bf;
typedef __attribute__((ext_vector_type(8)))  float  v8f;

#define BM 128
#define BN 128
#define BK 32
#define SA 40   // LDS row stride (elements) for A tile  [BM][BK] (+pad)
#define SB 40   // LDS row stride (elements) for B tile, stored transposed [BN][BK]

union FragU { uint4 q[2]; v16bf v; };            // 32 bytes
union Pack4 { unsigned short b[4]; uint2 u; };   // 8 bytes

// fp32 -> bf16 bits, round-to-nearest-even (inputs are finite)
static __device__ __forceinline__ unsigned short f2bf(float f) {
  unsigned u = __float_as_uint(f);
  unsigned r = u + 0x7FFFu + ((u >> 16) & 1u);
  return (unsigned short)(r >> 16);
}

// ---------------------------------------------------------------------------
// C[M,N] = A[M,K] * B[K,N] + bias[N]   (optionally relu), fp32 in/out,
// bf16 WMMA compute with fp32 accumulation. M%128==N%128==K%32==0.
// Software-pipelined: double-buffered LDS, global loads for tile k+1 issued
// before the WMMAs of tile k so s_wait_loadcnt lands after the matrix work.
// ---------------------------------------------------------------------------
__global__ __launch_bounds__(256) void gemm_bf16_kernel(
    const float* __restrict__ A, const float* __restrict__ B,
    const float* __restrict__ bias, float* __restrict__ C,
    int M, int N, int K, int relu)
{
  (void)M;
  __shared__ unsigned short As[2][BM * SA];   // As[b][m][k]
  __shared__ unsigned short Bs[2][BN * SB];   // Bs[b][n][k]  (transposed on fill)

  const int tid  = threadIdx.x;
  const int lane = tid & 31;
  const int w    = tid >> 5;        // wave 0..7
  const int wm   = (w & 3) * 32;    // wave M offset in tile
  const int wn   = (w >> 2) * 64;   // wave N offset in tile
  const int blockN = blockIdx.x * BN;
  const int blockM = blockIdx.y * BM;

  // global->LDS assignments
  const int a_k = (tid & 7) * 4;    // 0..28
  const int a_m = tid >> 3;         // 0..31 ; 4 rows each, stride 32
  const int b_n = (tid & 31) * 4;   // 0..124
  const int b_k = tid >> 5;         // 0..7  ; 4 rows each, stride 8

  // fragment addressing (ISA §7.12.2)
  const int arow = wm + (lane & 15);
  const int akb  = (lane >> 4) * 8;      // A: K chunk base {0,8}
  const int bkb  = (lane >> 4) * 8;      // B: K chunk base {0,8}
  const int ln   = lane & 15;

  v8f acc[2][4] = {};

  float4 ra[4], rb[4];

  // ---- prologue: fetch + stage tile k0 = 0 into buffer 0 ----
  #pragma unroll
  for (int r = 0; r < 4; ++r)
    ra[r] = *reinterpret_cast<const float4*>(
        A + (size_t)(blockM + a_m + r * 32) * K + a_k);
  #pragma unroll
  for (int r = 0; r < 4; ++r)
    rb[r] = *reinterpret_cast<const float4*>(
        B + (size_t)(b_k + r * 8) * N + (blockN + b_n));
  #pragma unroll
  for (int r = 0; r < 4; ++r) {
    Pack4 p;
    p.b[0] = f2bf(ra[r].x); p.b[1] = f2bf(ra[r].y);
    p.b[2] = f2bf(ra[r].z); p.b[3] = f2bf(ra[r].w);
    *reinterpret_cast<uint2*>(&As[0][(a_m + r * 32) * SA + a_k]) = p.u;
  }
  #pragma unroll
  for (int r = 0; r < 4; ++r) {
    const int k = b_k + r * 8;
    Bs[0][(b_n + 0) * SB + k] = f2bf(rb[r].x);
    Bs[0][(b_n + 1) * SB + k] = f2bf(rb[r].y);
    Bs[0][(b_n + 2) * SB + k] = f2bf(rb[r].z);
    Bs[0][(b_n + 3) * SB + k] = f2bf(rb[r].w);
  }
  __syncthreads();

  int buf = 0;
  for (int k0 = 0; k0 < K; k0 += BK) {
    const bool more = (k0 + BK) < K;

    // ---- issue next tile's global loads (consumed after the WMMAs) ----
    if (more) {
      const int kn = k0 + BK;
      #pragma unroll
      for (int r = 0; r < 4; ++r)
        ra[r] = *reinterpret_cast<const float4*>(
            A + (size_t)(blockM + a_m + r * 32) * K + (kn + a_k));
      #pragma unroll
      for (int r = 0; r < 4; ++r)
        rb[r] = *reinterpret_cast<const float4*>(
            B + (size_t)(kn + b_k + r * 8) * N + (blockN + b_n));
      if (k0 + 2 * BK < K) {  // warm L2 two tiles ahead (global_prefetch_b8)
        __builtin_prefetch(A + (size_t)(blockM + a_m) * K + (k0 + 2 * BK + a_k), 0, 0);
        __builtin_prefetch(B + (size_t)(k0 + 2 * BK + b_k) * N + (blockN + b_n), 0, 0);
      }
    }

    // ---- load all fragments, then 8 back-to-back WMMAs ----
    {
      const unsigned short* Ab = As[buf];
      const unsigned short* Bb = Bs[buf];
      FragU a0, a1, bfr[4];
      a0.q[0] = *reinterpret_cast<const uint4*>(&Ab[arow * SA + akb]);
      a0.q[1] = *reinterpret_cast<const uint4*>(&Ab[arow * SA + 16 + akb]);
      a1.q[0] = *reinterpret_cast<const uint4*>(&Ab[(arow + 16) * SA + akb]);
      a1.q[1] = *reinterpret_cast<const uint4*>(&Ab[(arow + 16) * SA + 16 + akb]);
      #pragma unroll
      for (int j = 0; j < 4; ++j) {
        const int bcol = wn + j * 16 + ln;
        bfr[j].q[0] = *reinterpret_cast<const uint4*>(&Bb[bcol * SB + bkb]);
        bfr[j].q[1] = *reinterpret_cast<const uint4*>(&Bb[bcol * SB + 16 + bkb]);
      }
      #pragma unroll
      for (int j = 0; j < 4; ++j) {
        acc[0][j] = __builtin_amdgcn_wmma_f32_16x16x32_bf16(
            false, a0.v, false, bfr[j].v, (short)0, acc[0][j], false, false);
        acc[1][j] = __builtin_amdgcn_wmma_f32_16x16x32_bf16(
            false, a1.v, false, bfr[j].v, (short)0, acc[1][j], false, false);
      }
    }

    // ---- convert + stage next tile into the other buffer ----
    if (more) {
      const int nb = buf ^ 1;
      #pragma unroll
      for (int r = 0; r < 4; ++r) {
        Pack4 p;
        p.b[0] = f2bf(ra[r].x); p.b[1] = f2bf(ra[r].y);
        p.b[2] = f2bf(ra[r].z); p.b[3] = f2bf(ra[r].w);
        *reinterpret_cast<uint2*>(&As[nb][(a_m + r * 32) * SA + a_k]) = p.u;
      }
      #pragma unroll
      for (int r = 0; r < 4; ++r) {
        const int k = b_k + r * 8;
        Bs[nb][(b_n + 0) * SB + k] = f2bf(rb[r].x);
        Bs[nb][(b_n + 1) * SB + k] = f2bf(rb[r].y);
        Bs[nb][(b_n + 2) * SB + k] = f2bf(rb[r].z);
        Bs[nb][(b_n + 3) * SB + k] = f2bf(rb[r].w);
      }
    }
    __syncthreads();
    buf ^= 1;
  }

  // ---- epilogue: C/D layout: VGPR r -> M = r + 8*(lane>=16), N = lane&15 ----
  const int hi = (lane >> 4) * 8;
  #pragma unroll
  for (int j = 0; j < 4; ++j) {
    const int ncol = blockN + wn + j * 16 + ln;
    const float bval = bias[ncol];
    #pragma unroll
    for (int i = 0; i < 2; ++i) {
      #pragma unroll
      for (int r = 0; r < 8; ++r) {
        const int mrow = blockM + wm + i * 16 + hi + r;
        float v = acc[i][j][r] + bval;
        if (relu) v = fmaxf(v, 0.f);
        C[(size_t)mrow * N + ncol] = v;
      }
    }
  }
}

// ---------------------------------------------------------------------------
// Wv repack: out[d*768 + h*64 + e] = Wv[h][d][e]   (one transformer block)
// ---------------------------------------------------------------------------
__global__ __launch_bounds__(256) void repack_wv_kernel(
    const float* __restrict__ Wv, float* __restrict__ out)
{
  const int idx = blockIdx.x * 256 + threadIdx.x;   // < 768*768
  const int d  = idx / 768;
  const int he = idx - d * 768;
  const int h  = he >> 6;
  const int e  = he & 63;
  out[idx] = Wv[((size_t)h * 768 + d) * 64 + e];
}

// ---------------------------------------------------------------------------
// X = LayerNorm(X + R) over last dim 768 (no eps, no affine, population var)
// one block (256 threads) per row
// ---------------------------------------------------------------------------
__global__ __launch_bounds__(256) void add_ln_kernel(
    float* __restrict__ X, const float* __restrict__ R)
{
  __shared__ float red[256];
  const int row = blockIdx.x;
  const int tid = threadIdx.x;
  float v[3];
  float s = 0.f;
  #pragma unroll
  for (int t = 0; t < 3; ++t) {
    const int c = tid + t * 256;
    v[t] = X[(size_t)row * 768 + c] + R[(size_t)row * 768 + c];
    s += v[t];
  }
  red[tid] = s; __syncthreads();
  for (int off = 128; off > 0; off >>= 1) {
    if (tid < off) red[tid] += red[tid + off];
    __syncthreads();
  }
  const float mu = red[0] * (1.f / 768.f);
  __syncthreads();
  float s2 = 0.f;
  #pragma unroll
  for (int t = 0; t < 3; ++t) { const float d = v[t] - mu; s2 += d * d; }
  red[tid] = s2; __syncthreads();
  for (int off = 128; off > 0; off >>= 1) {
    if (tid < off) red[tid] += red[tid + off];
    __syncthreads();
  }
  const float inv = 1.f / sqrtf(red[0] * (1.f / 768.f));
  #pragma unroll
  for (int t = 0; t < 3; ++t) {
    const int c = tid + t * 256;
    X[(size_t)row * 768 + c] = (v[t] - mu) * inv;
  }
}

// ---------------------------------------------------------------------------
// in-place softmax over rows of length 8192; one block per row
// ---------------------------------------------------------------------------
__global__ __launch_bounds__(256) void softmax_rows_kernel(float* __restrict__ P)
{
  __shared__ float red[256];
  const int row = blockIdx.x;
  const int tid = threadIdx.x;
  float x[32];
  float mx = -3.402823466e+38f;
  #pragma unroll
  for (int t = 0; t < 32; ++t) {
    x[t] = P[(size_t)row * 8192 + tid + t * 256];
    mx = fmaxf(mx, x[t]);
  }
  red[tid] = mx; __syncthreads();
  for (int off = 128; off > 0; off >>= 1) {
    if (tid < off) red[tid] = fmaxf(red[tid], red[tid + off]);
    __syncthreads();
  }
  mx = red[0]; __syncthreads();
  float s = 0.f;
  #pragma unroll
  for (int t = 0; t < 32; ++t) { x[t] = expf(x[t] - mx); s += x[t]; }
  red[tid] = s; __syncthreads();
  for (int off = 128; off > 0; off >>= 1) {
    if (tid < off) red[tid] += red[tid + off];
    __syncthreads();
  }
  const float inv = 1.f / red[0];
  #pragma unroll
  for (int t = 0; t < 32; ++t)
    P[(size_t)row * 8192 + tid + t * 256] = x[t] * inv;
}

// ---------------------------------------------------------------------------
// driver
// ---------------------------------------------------------------------------
extern "C" void kernel_launch(void* const* d_in, const int* in_sizes, int n_in,
                              void* d_out, int out_size, void* d_ws, size_t ws_size,
                              hipStream_t stream)
{
  (void)in_sizes; (void)n_in; (void)out_size; (void)ws_size;

  const int M  = 4096;   // B*S
  const int D  = 768;
  const int HD = 3072;
  const int Vv = 8192;
  const int NB = 4;

  const float* Xin   = (const float*)d_in[0];
  const float* emb_w = (const float*)d_in[1];
  const float* emb_b = (const float*)d_in[2];
  // d_in[3..6] = Wq,bq,Wk,bk : mathematically unused (softmax rows sum to 1)
  const float* Wv    = (const float*)d_in[7];   // [NB,H,D,DK]
  const float* bv    = (const float*)d_in[8];   // [NB,H,DK] == [NB,768] contiguous
  const float* Wo    = (const float*)d_in[9];   // [NB,768,768]
  const float* bo    = (const float*)d_in[10];  // [NB,768]
  const float* W1    = (const float*)d_in[11];  // [NB,768,3072]
  const float* b1    = (const float*)d_in[12];  // [NB,3072]
  const float* W2    = (const float*)d_in[13];  // [NB,3072,768]
  const float* b2    = (const float*)d_in[14];  // [NB,768]
  const float* Wout  = (const float*)d_in[15];  // [768,8192]
  const float* bout  = (const float*)d_in[16];  // [8192]
  float* out = (float*)d_out;                   // [4096,8192]

  // workspace carve-up (~90 MB)
  float* Xa    = (float*)d_ws;                  // [4096,768]
  float* Bu1   = Xa  + (size_t)M * D;           // [4096,768]
  float* Bu2   = Bu1 + (size_t)M * D;           // [4096,768]
  float* Hid   = Bu2 + (size_t)M * D;           // [4096,3072]
  float* Wvcat = Hid + (size_t)M * HD;          // [768,768]

  const dim3 blk(256);

  // X = Xin @ emb_w + emb_b
  gemm_bf16_kernel<<<dim3(D / 128, M / 128), blk, 0, stream>>>(
      Xin, emb_w, emb_b, Xa, M, D, Vv, 0);

  for (int i = 0; i < NB; ++i) {
    // cat = X @ Wv_cat + bv   (attention collapses to the V projection)
    repack_wv_kernel<<<(D * D) / 256, blk, 0, stream>>>(
        Wv + (size_t)i * 12 * 768 * 64, Wvcat);
    gemm_bf16_kernel<<<dim3(D / 128, M / 128), blk, 0, stream>>>(
        Xa, Wvcat, bv + (size_t)i * D, Bu1, M, D, D, 0);
    // O = cat @ Wo + bo
    gemm_bf16_kernel<<<dim3(D / 128, M / 128), blk, 0, stream>>>(
        Bu1, Wo + (size_t)i * D * D, bo + (size_t)i * D, Bu2, M, D, D, 0);
    // X = LN(X + O)
    add_ln_kernel<<<M, blk, 0, stream>>>(Xa, Bu2);
    // H = relu(X @ W1 + b1)
    gemm_bf16_kernel<<<dim3(HD / 128, M / 128), blk, 0, stream>>>(
        Xa, W1 + (size_t)i * D * HD, b1 + (size_t)i * HD, Hid, M, HD, D, 1);
    // F = H @ W2 + b2
    gemm_bf16_kernel<<<dim3(D / 128, M / 128), blk, 0, stream>>>(
        Hid, W2 + (size_t)i * HD * D, b2 + (size_t)i * D, Bu2, M, D, HD, 0);
    // X = LN(X + F)
    add_ln_kernel<<<M, blk, 0, stream>>>(Xa, Bu2);
  }

  // logits = X @ Wout + bout ; out = softmax(logits)
  gemm_bf16_kernel<<<dim3(Vv / 128, M / 128), blk, 0, stream>>>(
      Xa, Wout, bout, out, M, Vv, D, 0);
  softmax_rows_kernel<<<M, blk, 0, stream>>>(out);
}
